// KroneckerGATorch_63204738728187
// MI455X (gfx1250) — matrix operand order repaired and checked by reading, not compiled, not run on hardware
//
#include <hip/hip_runtime.h>

typedef _Float16 v16h  __attribute__((ext_vector_type(16)));
typedef _Float16 half8 __attribute__((ext_vector_type(8)));
typedef _Float16 half2v __attribute__((ext_vector_type(2)));
typedef float    v8f   __attribute__((ext_vector_type(8)));

#define POP  4096
#define KDIM 8192   // 2 * POP : weight split into f16-exact hi + lo parts
#define OUTN 256

// ---------------------------------------------------------------------------
// Phase 1: per genome p build
//   A_p[a] = prod over codons 0..3 of 2x2 entries (a = iA*16+jA), values in {-1,0,1}
//   R_p[b] = prod over codons 4..7
// and store (K-major rows so the GEMM can do contiguous b128 fragment loads):
//   M1[a][2p]   = f16(w_hi[p]) * A_p[a]     (exact: A in {-1,0,1})
//   M1[a][2p+1] = f16(w_lo[p]) * A_p[a]
//   M2[b][2p]   = M2[b][2p+1] = f16(R_p[b]) (exact)
// ---------------------------------------------------------------------------
__global__ __launch_bounds__(256) void kga_prep(
    const float* __restrict__ w, const int* __restrict__ idx,
    const float* __restrict__ blocks, _Float16* __restrict__ M1,
    _Float16* __restrict__ M2)
{
  __shared__ float sblk[144];            // 36 blocks * 4 entries
  const int t = threadIdx.x;
  const int p = blockIdx.x;
  if (t < 144) sblk[t] = blocks[t];
  __syncthreads();

  // f16-exact two-term split of the weight: w ~= whi + wlo (f32-accurate)
  const float wv  = w[p];
  const float whi = (float)(_Float16)wv;
  const float wlo = (float)(_Float16)(wv - whi);

  const int i0 = idx[p*8+0], i1 = idx[p*8+1], i2 = idx[p*8+2], i3 = idx[p*8+3];
  const int i4 = idx[p*8+4], i5 = idx[p*8+5], i6 = idx[p*8+6], i7 = idx[p*8+7];

  const int a = t;  // a = iA*16 + jA ; bit(7-c) = codon-c row bit, bit(3-c) = col bit
  const float aval = sblk[i0*4 + ((a>>7)&1)*2 + ((a>>3)&1)]
                   * sblk[i1*4 + ((a>>6)&1)*2 + ((a>>2)&1)]
                   * sblk[i2*4 + ((a>>5)&1)*2 + ((a>>1)&1)]
                   * sblk[i3*4 + ((a>>4)&1)*2 + ( a     &1)];
  const float rval = sblk[i4*4 + ((a>>7)&1)*2 + ((a>>3)&1)]
                   * sblk[i5*4 + ((a>>6)&1)*2 + ((a>>2)&1)]
                   * sblk[i6*4 + ((a>>5)&1)*2 + ((a>>1)&1)]
                   * sblk[i7*4 + ((a>>4)&1)*2 + ( a     &1)];

  half2v m1; m1[0] = (_Float16)(whi * aval); m1[1] = (_Float16)(wlo * aval);
  half2v m2; m2[0] = (_Float16)rval;         m2[1] = (_Float16)rval;
  *(half2v*)(M1 + (size_t)a * KDIM + 2*p) = m1;
  *(half2v*)(M2 + (size_t)a * KDIM + 2*p) = m2;
}

// ---------------------------------------------------------------------------
// Phase 2: T = M1^T * M2  (256x256, K=8192) via v_wmma_f32_16x16x32_f16.
// One block (8 waves) per 16x16 tile; waves split K, LDS reduce, then
// permuted store:  Out[iA*16+iR, jA*16+jR] = T[iA*16+jA, iR*16+jR].
// ---------------------------------------------------------------------------
__global__ __launch_bounds__(256) void kga_gemm(
    const _Float16* __restrict__ M1, const _Float16* __restrict__ M2,
    float* __restrict__ out)
{
  __shared__ float red[8][32][8];        // [wave][lane][row-reg]  8KB
  const int lane = threadIdx.x & 31;
  const int wv   = threadIdx.x >> 5;
  const int tile = blockIdx.x;
  const int a0 = (tile >> 4) << 4;
  const int b0 = (tile & 15) << 4;

  const int mn = lane & 15;              // m for A fragment, n for B fragment
  const int g  = lane >> 4;              // lane-half selects K sub-group

  // ISA 7.12.2 fragment K maps (relative to k0):
  //   A (16x32 f16): e<8 -> K = e + 8g ;  e>=8 -> K = e+8 + 8g
  //   B (32x16 f16): e<8 -> K = e + 8g ;  e>=8 -> K = e+8 + 8g
  const _Float16* rowA = M1 + (size_t)(a0 + mn) * KDIM + g * 8;
  const _Float16* rowB = M2 + (size_t)(b0 + mn) * KDIM + g * 8;

  v8f acc = {};
  const int kbeg = wv * (KDIM / 8);      // 1024 K per wave -> 32 WMMAs
  #pragma unroll 2
  for (int k0 = kbeg; k0 < kbeg + KDIM / 8; k0 += 32) {
    half8 alo = *(const half8*)(rowA + k0);
    half8 ahi = *(const half8*)(rowA + k0 + 16);
    half8 blo = *(const half8*)(rowB + k0);
    half8 bhi = *(const half8*)(rowB + k0 + 16);
    v16h av, bvv;
    #pragma unroll
    for (int e = 0; e < 8; ++e) {
      av[e] = alo[e]; av[e + 8] = ahi[e];
      bvv[e] = blo[e]; bvv[e + 8] = bhi[e];
    }
    acc = __builtin_amdgcn_wmma_f32_16x16x32_f16(
        /*neg_a=*/false, av, /*neg_b=*/false, bvv,
        /*c_mod=*/(short)0, acc, /*reuse_a=*/false, /*reuse_b=*/false);
  }

  #pragma unroll
  for (int r = 0; r < 8; ++r) red[wv][lane][r] = acc[r];
  __syncthreads();

  // thread t owns tile element (m,n); C/D layout: lane = n + 16*(m>=8), reg = m&7
  const int m = threadIdx.x >> 4;
  const int n = threadIdx.x & 15;
  const int slane = n + ((m >= 8) ? 16 : 0);
  const int sr = m & 7;
  float s = 0.f;
  #pragma unroll
  for (int q = 0; q < 8; ++q) s += red[q][slane][sr];

  const int aa = a0 + m, bb = b0 + n;
  const int i = ((aa >> 4) << 4) | (bb >> 4);   // iA*16 + iR
  const int j = ((aa & 15) << 4) | (bb & 15);   // jA*16 + jR
  out[i * OUTN + j] = s;
}

extern "C" void kernel_launch(void* const* d_in, const int* in_sizes, int n_in,
                              void* d_out, int out_size, void* d_ws, size_t ws_size,
                              hipStream_t stream) {
  (void)in_sizes; (void)n_in; (void)out_size; (void)ws_size;
  const float* w      = (const float*)d_in[0];   // [4096] f32
  const int*   idx    = (const int*)  d_in[1];   // [4096,8] i32
  const float* blocks = (const float*)d_in[2];   // [36,2,2] f32
  float* out = (float*)d_out;                    // [256,256] f32

  _Float16* M1 = (_Float16*)d_ws;                // 256 x 8192 f16 (4 MB)
  _Float16* M2 = M1 + (size_t)OUTN * KDIM;       // 256 x 8192 f16 (4 MB)

  kga_prep<<<POP, 256, 0, stream>>>(w, idx, blocks, M1, M2);
  kga_gemm<<<OUTN, 256, 0, stream>>>(M1, M2, out);
}
